// encoder_attention_5411658793185
// MI455X (gfx1250) — matrix operand order repaired and checked by reading, not compiled, not run on hardware
//
#include <hip/hip_runtime.h>
#include <math.h>

// ---------------- problem constants ----------------
#define BB   32
#define LL   4096
#define H2   512
#define EE   128
#define ROWS (BB*LL)            // 131072 M rows
#define CT_SIZE (BB*H2)         // 16384
#define AT_OFF  CT_SIZE
#define NS_OFF  (CT_SIZE + BB*LL)

// workspace layout (bytes)
#define WS_WHI  0                       // 512 frags * 32 lanes * 8 dwords = 512KB
#define WS_WLO  (512*1024)              // 512KB
#define WS_BIAS (1024*1024)             // 64KB (B*H2 floats)
#define WS_RSUM (1024*1024 + 64*1024)   // 32 floats

typedef __attribute__((ext_vector_type(16))) __bf16  v16bf;
typedef __attribute__((ext_vector_type(8)))  float   v8f;
typedef __attribute__((ext_vector_type(8)))  unsigned int v8u;

__device__ __forceinline__ unsigned short f2bf(float f) {
    unsigned int u = __float_as_uint(f);
    unsigned int r = (u + 0x7FFFu + ((u >> 16) & 1u)) >> 16;   // RNE
    return (unsigned short)r;
}
__device__ __forceinline__ float bf2f(unsigned short s) {
    return __uint_as_float(((unsigned int)s) << 16);
}

// ---------------------------------------------------------------------------
// Kernel 1: bias[b][e] = W_s(st_hat) + W_s_b + W_t(sum_k_emb) + W_t_b
// ---------------------------------------------------------------------------
__global__ __launch_bounds__(256) void bias_kernel(
    const float* __restrict__ st_hat, const float* __restrict__ ske,
    const float* __restrict__ Wsw, const float* __restrict__ Wsb,
    const float* __restrict__ Wtw, const float* __restrict__ Wtb,
    float* __restrict__ bias)
{
    int i = blockIdx.x * 256 + threadIdx.x;      // 0 .. B*H2-1
    int e = i & (H2 - 1);
    int b = i >> 9;
    float acc = Wsb[e] + Wtb[e];
    const float* ws = Wsw + e * H2;
    const float* sh = st_hat + b * H2;
    #pragma unroll 4
    for (int d = 0; d < H2; ++d) acc += sh[d] * ws[d];
    const float* wt = Wtw + e * EE;
    const float* sk = ske + b * EE;
    #pragma unroll 4
    for (int j = 0; j < EE; ++j) acc += sk[j] * wt[j];
    bias[i] = acc;
}

// ---------------------------------------------------------------------------
// Kernel 2: W_h -> bf16 hi/lo split, pre-swizzled into WMMA B-fragment order.
// Fragment (ntile, kstep): per lane (half = lane>>4, n = ntile*16 + lane&15),
// dword v (0..7) holds bf16 pair at k = kstep*32 + 16*(v>>2) + 8*half + 2*(v&3).
// B[k][n] = W_h[n][k]  (since C[m,e] = sum_d h[m,d] * W_h[e,d]).
// ---------------------------------------------------------------------------
__global__ __launch_bounds__(256) void wprep_kernel(
    const float* __restrict__ Wh,
    unsigned int* __restrict__ whi, unsigned int* __restrict__ wlo)
{
    int idx  = blockIdx.x * 256 + threadIdx.x;   // 0 .. 131071
    int v    = idx & 7;
    int lane = (idx >> 3) & 31;
    int frag = idx >> 8;                          // ntile*16 + kstep
    int ks   = frag & 15;
    int nt   = frag >> 4;
    int hf   = lane >> 4;
    int n    = nt * 16 + (lane & 15);
    int k    = ks * 32 + 16 * (v >> 2) + 8 * hf + 2 * (v & 3);
    float x0 = Wh[n * H2 + k];
    float x1 = Wh[n * H2 + k + 1];
    unsigned short h0 = f2bf(x0), h1 = f2bf(x1);
    whi[idx] = ((unsigned)h1 << 16) | h0;
    unsigned short l0 = f2bf(x0 - bf2f(h0)), l1 = f2bf(x1 - bf2f(h1));
    wlo[idx] = ((unsigned)l1 << 16) | l0;
}

// ---------------------------------------------------------------------------
// Kernel 3 (main): scores via bf16x3 WMMA, fused tanh + dot(v_w) + exp.
// 4 waves/block, each wave owns a 16-row M-tile. A-tile (16x512) converted
// once to hi/lo bf16 in LDS in A-fragment order, reused across 32 N-tiles.
// Writes unnormalized at and new_sum_temporal.
// ---------------------------------------------------------------------------
__global__ __launch_bounds__(128) void scores_kernel(
    const float* __restrict__ h, const float* __restrict__ mask,
    const float* __restrict__ sumtmp, const float* __restrict__ vw,
    const float* __restrict__ bias,
    const unsigned int* __restrict__ whi, const unsigned int* __restrict__ wlo,
    float* __restrict__ out)
{
    extern __shared__ unsigned int smem[];       // 4 waves * (4096 hi + 4096 lo) dwords
    const int lane = threadIdx.x & 31;
    const int wv   = threadIdx.x >> 5;
    const int half = lane >> 4;
    const int l15  = lane & 15;
    const int rowbase = blockIdx.x * 64 + wv * 16;   // 16 consecutive (b,l) rows
    const int bIdx = rowbase >> 12;                  // L = 4096

    unsigned int* Ahi = smem + wv * 8192;
    unsigned int* Alo = Ahi + 4096;

    // ---- stage: convert 16x512 fp32 A-tile into hi/lo bf16 A-fragments ----
    #pragma unroll 4
    for (int i = 0; i < 128; ++i) {
        int idx = i * 32 + lane;        // pair-dword id, 0..4095
        int row = idx >> 8;             // 0..15
        int p   = idx & 255;
        int k   = p * 2;
        const float2 x = *(const float2*)(h + (size_t)(rowbase + row) * H2 + k);
        int ks   = k >> 5;
        int w32  = k & 31;
        int hf   = (w32 >> 3) & 1;
        int base = w32 - 8 * hf;        // in {0,2,4,6,16,18,20,22}
        int v    = (base < 8) ? (base >> 1) : (4 + ((base - 16) >> 1));
        int daddr = (ks * 32 + (row + 16 * hf)) * 8 + v;
        unsigned short h0 = f2bf(x.x), h1 = f2bf(x.y);
        Ahi[daddr] = ((unsigned)h1 << 16) | h0;
        unsigned short l0 = f2bf(x.x - bf2f(h0)), l1 = f2bf(x.y - bf2f(h1));
        Alo[daddr] = ((unsigned)l1 << 16) | l0;
    }
    __syncthreads();

    v8f zero8;
    #pragma unroll
    for (int q = 0; q < 8; ++q) zero8[q] = 0.0f;

    float sc[8];
    #pragma unroll
    for (int r = 0; r < 8; ++r) sc[r] = 0.0f;

    // ---- main loop: 8 chunks of 4 N-tiles, 16 K-steps, 3 WMMAs each ----
    for (int nc = 0; nc < 8; ++nc) {
        v8f c[4];
        #pragma unroll
        for (int t = 0; t < 4; ++t) c[t] = zero8;

        for (int ks = 0; ks < 16; ++ks) {
            const int aoff = (ks * 32 + lane) * 8;
            v8u ahu = *(const v8u*)(Ahi + aoff);
            v8u alu = *(const v8u*)(Alo + aoff);
            v16bf ahv = __builtin_bit_cast(v16bf, ahu);
            v16bf alv = __builtin_bit_cast(v16bf, alu);
            #pragma unroll
            for (int t = 0; t < 4; ++t) {
                int nt = nc * 4 + t;
                size_t fo = ((size_t)(nt * 16 + ks) * 32 + lane) * 8;
                v8u bhu = *(const v8u*)(whi + fo);
                v8u blu = *(const v8u*)(wlo + fo);
                v16bf bhv = __builtin_bit_cast(v16bf, bhu);
                v16bf blv = __builtin_bit_cast(v16bf, blu);
                c[t] = __builtin_amdgcn_wmma_f32_16x16x32_bf16(
                           false, ahv, false, bhv, (short)0, c[t], false, false);
                c[t] = __builtin_amdgcn_wmma_f32_16x16x32_bf16(
                           false, ahv, false, blv, (short)0, c[t], false, false);
                c[t] = __builtin_amdgcn_wmma_f32_16x16x32_bf16(
                           false, alv, false, bhv, (short)0, c[t], false, false);
            }
        }
        // finalize 4 N-tiles: tanh(et + bias) * v_w, accumulate per-lane
        #pragma unroll
        for (int t = 0; t < 4; ++t) {
            int e = (nc * 4 + t) * 16 + l15;
            float bb = bias[bIdx * H2 + e];
            float vv = vw[e];
            #pragma unroll
            for (int r = 0; r < 8; ++r)
                sc[r] += tanhf(c[t][r] + bb) * vv;
        }
    }

    // ---- reduce over the 16 N-lanes within each half ----
    #pragma unroll
    for (int r = 0; r < 8; ++r) {
        for (int off = 8; off > 0; off >>= 1)
            sc[r] += __shfl_xor(sc[r], off, 16);
    }

    // lanes 0 (M rows 0..7) and 16 (M rows 8..15) hold the scores
    if (l15 == 0) {
        int base = rowbase + 8 * half;
        #pragma unroll
        for (int r = 0; r < 8; ++r) {
            int m = base + r;
            float ex   = expf(sc[r]);
            float prev = sumtmp[m];
            out[NS_OFF + m] = prev + ex;                 // new_sum_temporal
            out[AT_OFF + m] = (ex / prev) * mask[m];     // unnormalized at
        }
    }
}

// ---------------------------------------------------------------------------
// Kernel 4: deterministic per-batch row sum of unnormalized at
// ---------------------------------------------------------------------------
__global__ __launch_bounds__(256) void rowsum_kernel(
    const float* __restrict__ atun, float* __restrict__ rowsum)
{
    __shared__ float red[256];
    int b = blockIdx.x, t = threadIdx.x;
    float s = 0.0f;
    for (int l = t; l < LL; l += 256) s += atun[b * LL + l];
    red[t] = s;
    __syncthreads();
    for (int o = 128; o > 0; o >>= 1) {
        if (t < o) red[t] += red[t + o];
        __syncthreads();
    }
    if (t == 0) rowsum[b] = red[0];
}

// ---------------------------------------------------------------------------
// Kernel 5: normalize at in place
// ---------------------------------------------------------------------------
__global__ __launch_bounds__(256) void atnorm_kernel(
    float* __restrict__ at, const float* __restrict__ rowsum)
{
    int i = blockIdx.x * 256 + threadIdx.x;      // 0 .. B*L-1
    int b = i >> 12;
    at[i] = at[i] / rowsum[b];
}

// ---------------------------------------------------------------------------
// Kernel 6: ct_e[b][d] = sum_l at[b][l] * h[b][l][d]   (bandwidth bound)
// ---------------------------------------------------------------------------
__global__ __launch_bounds__(256) void ct_kernel(
    const float* __restrict__ h, const float* __restrict__ atn,
    float* __restrict__ ct)
{
    int b = blockIdx.x >> 1;
    int d = ((blockIdx.x & 1) * 256) + threadIdx.x;
    const float* hb = h + (size_t)b * LL * H2 + d;
    const float* ab = atn + b * LL;
    float acc = 0.0f;
    for (int l = 0; l < LL; l += 4) {
        acc += ab[l + 0] * hb[(size_t)(l + 0) * H2];
        acc += ab[l + 1] * hb[(size_t)(l + 1) * H2];
        acc += ab[l + 2] * hb[(size_t)(l + 2) * H2];
        acc += ab[l + 3] * hb[(size_t)(l + 3) * H2];
    }
    ct[b * H2 + d] = acc;
}

// ---------------------------------------------------------------------------
extern "C" void kernel_launch(void* const* d_in, const int* in_sizes, int n_in,
                              void* d_out, int out_size, void* d_ws, size_t ws_size,
                              hipStream_t stream) {
    (void)in_sizes; (void)n_in; (void)out_size; (void)ws_size;
    const float* st_hat = (const float*)d_in[0];
    const float* h      = (const float*)d_in[1];
    const float* mask   = (const float*)d_in[2];
    const float* sumt   = (const float*)d_in[3];
    const float* ske    = (const float*)d_in[4];
    const float* Wh     = (const float*)d_in[5];
    const float* Wsw    = (const float*)d_in[6];
    const float* Wsb    = (const float*)d_in[7];
    const float* Wtw    = (const float*)d_in[8];
    const float* Wtb    = (const float*)d_in[9];
    const float* vw     = (const float*)d_in[10];
    float* out = (float*)d_out;
    char*  ws  = (char*)d_ws;

    unsigned int* whi    = (unsigned int*)(ws + WS_WHI);
    unsigned int* wlo    = (unsigned int*)(ws + WS_WLO);
    float*        bias   = (float*)(ws + WS_BIAS);
    float*        rowsum = (float*)(ws + WS_RSUM);

    bias_kernel<<<(BB * H2) / 256, 256, 0, stream>>>(st_hat, ske, Wsw, Wsb, Wtw, Wtb, bias);
    wprep_kernel<<<512, 256, 0, stream>>>(Wh, whi, wlo);
    scores_kernel<<<ROWS / 64, 128, 128 * 1024, stream>>>(h, mask, sumt, vw, bias,
                                                          whi, wlo, out);
    rowsum_kernel<<<BB, 256, 0, stream>>>(out + AT_OFF, rowsum);
    atnorm_kernel<<<(BB * LL) / 256, 256, 0, stream>>>(out + AT_OFF, rowsum);
    ct_kernel<<<BB * 2, 256, 0, stream>>>(h, out + AT_OFF, out);
}
